// CausalSelfAttention_TSSA_86870008529343
// MI455X (gfx1250) — compile-verified, matrix operand order checked by reading, not run on hardware
//
#include <hip/hip_runtime.h>
#include <hip/hip_bf16.h>

// ---------------------------------------------------------------------------
// CDNA5 (gfx1250) implementation of the TSSA causal-attention reference.
//   GEMMs run on v_wmma_f32_16x16x32_bf16 with an error-compensated
//   bf16 hi/lo split (hi*hi + hi*lo + lo*hi) to keep ~fp32 accuracy.
//   Cumsum/softmax stages are register-resident sequential scans.
// ---------------------------------------------------------------------------

typedef __attribute__((ext_vector_type(16))) __bf16 v16bf;
typedef __attribute__((ext_vector_type(8)))  float  v8f;

union Frag16 {
    v16bf v;
    uint4 u[2];
};

__device__ __forceinline__ unsigned short f2bf(float f) {
    // round-to-nearest-even bf16 truncation
    unsigned u = __float_as_uint(f);
    unsigned r = 0x7fffu + ((u >> 16) & 1u);
    return (unsigned short)((u + r) >> 16);
}
__device__ __forceinline__ float bf2f(unsigned short h) {
    return __uint_as_float(((unsigned)h) << 16);
}

// ---------------------------------------------------------------------------
// Split f32 -> bf16 hi + bf16 lo residual
// ---------------------------------------------------------------------------
__global__ void split_bf16(const float* __restrict__ in,
                           unsigned short* __restrict__ hi,
                           unsigned short* __restrict__ lo, int n) {
    int i = blockIdx.x * blockDim.x + threadIdx.x;
    if (i >= n) return;
    float f = in[i];
    unsigned short h = f2bf(f);
    hi[i] = h;
    lo[i] = f2bf(f - bf2f(h));
}

// ---------------------------------------------------------------------------
// C = A @ B^T + bias, via v_wmma_f32_16x16x32_bf16 with hi/lo compensation.
// A: (M,K) row-major (hi/lo bf16).  Brows: (N,K) row-major (hi/lo bf16),
// i.e. B^T columns are Brows rows -> WMMA B fragment loads are contiguous.
// Each wave computes a 16x64 strip (4 tiles sharing the A fragment).
// ---------------------------------------------------------------------------
__device__ __forceinline__ Frag16 load_frag(const unsigned short* __restrict__ base,
                                            int K, int row, int k, int koff) {
    Frag16 f;
    const unsigned short* p = base + (size_t)row * K + k + koff;
    f.u[0] = *(const uint4*)p;         // K chunk 0..7  (lane<16) / 8..15 (lane>=16)
    f.u[1] = *(const uint4*)(p + 16);  // K chunk 16..23 / 24..31
    return f;
}

__global__ __launch_bounds__(256)
void wmma_gemm_bias(const unsigned short* __restrict__ Ahi,
                    const unsigned short* __restrict__ Alo,
                    const unsigned short* __restrict__ Bhi,
                    const unsigned short* __restrict__ Blo,
                    const float* __restrict__ bias,
                    float* __restrict__ Cout,
                    int M, int N, int K) {
    const int lane    = threadIdx.x & 31;
    const int wave    = blockIdx.x * (blockDim.x >> 5) + (threadIdx.x >> 5);
    const int nstrips = N >> 6;                 // strips of 64 columns
    const int mtile   = wave / nstrips;
    const int nstrip  = wave % nstrips;
    if ((mtile << 4) >= M) return;

    const int m0   = mtile << 4;
    const int n0   = nstrip << 6;
    const int rowA = m0 + (lane & 15);
    const int koff = (lane >> 4) << 3;          // 0 or 8 (ISA 16-bit A layout)

    v8f zero = {};
    v8f acc[4];
    acc[0] = zero; acc[1] = zero; acc[2] = zero; acc[3] = zero;

    for (int k = 0; k < K; k += 32) {
        Frag16 ah = load_frag(Ahi, K, rowA, k, koff);
        Frag16 al = load_frag(Alo, K, rowA, k, koff);
        // prefetch next K-chunk of the A stream (global_prefetch_b8)
        if (k + 32 < K) {
            __builtin_prefetch(Ahi + (size_t)rowA * K + k + 32 + koff, 0, 1);
        }
#pragma unroll
        for (int j = 0; j < 4; ++j) {
            const int rowB = n0 + (j << 4) + (lane & 15);
            Frag16 bh = load_frag(Bhi, K, rowB, k, koff);
            Frag16 bl = load_frag(Blo, K, rowB, k, koff);
            acc[j] = __builtin_amdgcn_wmma_f32_16x16x32_bf16(
                false, ah.v, false, bh.v, (short)0, acc[j], false, false);
            acc[j] = __builtin_amdgcn_wmma_f32_16x16x32_bf16(
                false, ah.v, false, bl.v, (short)0, acc[j], false, false);
            acc[j] = __builtin_amdgcn_wmma_f32_16x16x32_bf16(
                false, al.v, false, bh.v, (short)0, acc[j], false, false);
        }
    }

    // C/D layout: VGPR v holds M = m0 + v + 8*(lane>=16), N = n0 + lane%16
    const int ncol  = lane & 15;
    const int mbase = m0 + ((lane >> 4) << 3);
#pragma unroll
    for (int j = 0; j < 4; ++j) {
        const int n  = n0 + (j << 4) + ncol;
        const float bv = bias[n];
#pragma unroll
        for (int v = 0; v < 8; ++v) {
            Cout[(size_t)(mbase + v) * N + n] = acc[j][v] + bv;
        }
    }
}

// ---------------------------------------------------------------------------
// Stage 1: denom = cumsum_T(w^2); tmp[b,h,t] = (sum_d w^2/denom + D*dbias)*temp
// 16 lanes per (b,h); each lane owns 4 d-channels (float4).
// ---------------------------------------------------------------------------
__global__ void stage_tmp(const float* __restrict__ w,
                          const float* __restrict__ temp,
                          const float* __restrict__ denom_bias,
                          float* __restrict__ tmpout,
                          int B, int T, int C, int H, int D) {
    int g = (blockIdx.x * blockDim.x + threadIdx.x) >> 4;
    int l = threadIdx.x & 15;
    if (g >= B * H) return;
    int b = g / H, h = g % H;

    const float* base  = w + (size_t)b * T * C + h * D + l * 4;
    const float* dbias = denom_bias + (size_t)h * T;
    float* out = tmpout + (size_t)g * T;
    const float th = temp[h];

    float4 run = make_float4(0.f, 0.f, 0.f, 0.f);
    for (int t = 0; t < T; ++t) {
        float4 wv = *(const float4*)(base + (size_t)t * C);
        float sx = wv.x * wv.x, sy = wv.y * wv.y, sz = wv.z * wv.z, sw = wv.w * wv.w;
        run.x += sx; run.y += sy; run.z += sz; run.w += sw;
        float val = sx / fmaxf(run.x, 1e-12f) + sy / fmaxf(run.y, 1e-12f)
                  + sz / fmaxf(run.z, 1e-12f) + sw / fmaxf(run.w, 1e-12f);
        // reduce over the 16-lane group (stays within half-wave)
        val += __shfl_xor(val, 8, 32);
        val += __shfl_xor(val, 4, 32);
        val += __shfl_xor(val, 2, 32);
        val += __shfl_xor(val, 1, 32);
        if (l == 0) out[t] = (val + (float)D * dbias[t]) * th;
    }
}

// ---------------------------------------------------------------------------
// Stage 2: Pi = softmax over heads (axis H=16). One thread per (b,t).
// ---------------------------------------------------------------------------
__global__ void stage_softmax(const float* __restrict__ tmpin,
                              float* __restrict__ Pi, int B, int T, int H) {
    int idx = blockIdx.x * blockDim.x + threadIdx.x;
    if (idx >= B * T) return;
    int b = idx / T, t = idx % T;
    const float* p = tmpin + (size_t)b * H * T + t;
    float v[16];
    float m = -3.4e38f;
#pragma unroll
    for (int h = 0; h < 16; ++h) { v[h] = p[(size_t)h * T]; m = fmaxf(m, v[h]); }
    float s = 0.f;
#pragma unroll
    for (int h = 0; h < 16; ++h) { v[h] = __expf(v[h] - m); s += v[h]; }
    float inv = 1.f / s;
    float* q = Pi + (size_t)b * H * T + t;
#pragma unroll
    for (int h = 0; h < 16; ++h) q[(size_t)h * T] = v[h] * inv;
}

// ---------------------------------------------------------------------------
// Stage 3: dots = cumsum_T(w^2*Pi)/(cumsum_T(Pi)+1e-8); y = -(w*Pi)/(1+dots)
// Writes y straight into bf16 hi/lo buffers (GEMM2 inputs).
// ---------------------------------------------------------------------------
__global__ void stage_y(const float* __restrict__ w,
                        const float* __restrict__ Pi,
                        unsigned short* __restrict__ yhi,
                        unsigned short* __restrict__ ylo,
                        int B, int T, int C, int H, int D) {
    int g = (blockIdx.x * blockDim.x + threadIdx.x) >> 4;
    int l = threadIdx.x & 15;
    if (g >= B * H) return;
    int b = g / H, h = g % H;

    const float* wbase = w + (size_t)b * T * C + h * D + l * 4;
    const float* pbase = Pi + (size_t)g * T;
    size_t obase = (size_t)b * T * C + h * D + l * 4;

    float4 cumWP = make_float4(0.f, 0.f, 0.f, 0.f);
    float cumPi = 0.f;
    for (int t = 0; t < T; ++t) {
        float pi = pbase[t];
        cumPi += pi;
        float4 wv = *(const float4*)(wbase + (size_t)t * C);
        cumWP.x += wv.x * wv.x * pi;
        cumWP.y += wv.y * wv.y * pi;
        cumWP.z += wv.z * wv.z * pi;
        cumWP.w += wv.w * wv.w * pi;
        float dn = cumPi + 1e-8f;
        float yx = -(wv.x * pi) / (1.f + cumWP.x / dn);
        float yy = -(wv.y * pi) / (1.f + cumWP.y / dn);
        float yz = -(wv.z * pi) / (1.f + cumWP.z / dn);
        float yw = -(wv.w * pi) / (1.f + cumWP.w / dn);

        unsigned short h0 = f2bf(yx), h1 = f2bf(yy), h2 = f2bf(yz), h3 = f2bf(yw);
        unsigned short l0 = f2bf(yx - bf2f(h0)), l1 = f2bf(yy - bf2f(h1));
        unsigned short l2 = f2bf(yz - bf2f(h2)), l3 = f2bf(yw - bf2f(h3));

        size_t o = obase + (size_t)t * C;
        uint2 hv, lv;
        hv.x = (unsigned)h0 | ((unsigned)h1 << 16);
        hv.y = (unsigned)h2 | ((unsigned)h3 << 16);
        lv.x = (unsigned)l0 | ((unsigned)l1 << 16);
        lv.y = (unsigned)l2 | ((unsigned)l3 << 16);
        *(uint2*)(yhi + o) = hv;
        *(uint2*)(ylo + o) = lv;
    }
}

// ---------------------------------------------------------------------------
// Host-side launch
// ---------------------------------------------------------------------------
extern "C" void kernel_launch(void* const* d_in, const int* in_sizes, int n_in,
                              void* d_out, int out_size, void* d_ws, size_t ws_size,
                              hipStream_t stream) {
    const float* x          = (const float*)d_in[0];
    const float* Wa         = (const float*)d_in[1];
    const float* ba         = (const float*)d_in[2];
    const float* Wp         = (const float*)d_in[3];
    const float* bp         = (const float*)d_in[4];
    const float* temp       = (const float*)d_in[5];
    const float* denom_bias = (const float*)d_in[6];

    const int B = 4, T = 4096, C = 1024, H = 16, D = 64;
    const int M = B * T;   // 16384
    const int N = C, K = C;

    // Workspace layout (bytes)
    char* ws = (char*)d_ws;
    float*          wbuf = (float*)(ws + 0);                       // 64 MB  (B,T,C) f32
    unsigned short* hi   = (unsigned short*)(ws + 67108864);       // 32 MB  x-hi / y-hi
    unsigned short* lo   = (unsigned short*)(ws + 100663296);      // 32 MB  x-lo / y-lo
    unsigned short* Wahi = (unsigned short*)(ws + 134217728);      //  2 MB
    unsigned short* Walo = (unsigned short*)(ws + 136314880);      //  2 MB
    unsigned short* Wphi = (unsigned short*)(ws + 138412032);      //  2 MB
    unsigned short* Wplo = (unsigned short*)(ws + 140509184);      //  2 MB
    float*          tmpb = (float*)(ws + 142606336);               //  1 MB  (B,H,T)
    float*          Pib  = (float*)(ws + 143654912);               //  1 MB  (B,H,T)

    const int nX = M * C;      // 16.7M
    const int nW = C * C;      // 1M
    split_bf16<<<(nX + 255) / 256, 256, 0, stream>>>(x,  hi,   lo,   nX);
    split_bf16<<<(nW + 255) / 256, 256, 0, stream>>>(Wa, Wahi, Walo, nW);
    split_bf16<<<(nW + 255) / 256, 256, 0, stream>>>(Wp, Wphi, Wplo, nW);

    const int waves  = (M / 16) * (N / 64);   // 16384 waves
    const int blocks = waves / 8;             // 8 waves (256 thr) per block

    // GEMM1: w = x @ Wa.T + ba
    wmma_gemm_bias<<<blocks, 256, 0, stream>>>(hi, lo, Wahi, Walo, ba, wbuf, M, N, K);

    // Scan / softmax / scan
    stage_tmp<<<(B * H * 16 + 255) / 256, 256, 0, stream>>>(wbuf, temp, denom_bias, tmpb,
                                                            B, T, C, H, D);
    stage_softmax<<<(B * T + 255) / 256, 256, 0, stream>>>(tmpb, Pib, B, T, H);
    stage_y<<<(B * H * 16 + 255) / 256, 256, 0, stream>>>(wbuf, Pib, hi, lo,
                                                          B, T, C, H, D);

    // GEMM2: out = y @ Wp.T + bp
    wmma_gemm_bias<<<blocks, 256, 0, stream>>>(hi, lo, Wphi, Wplo, bp,
                                               (float*)d_out, M, N, K);
}
